// MultiHeadAttention_53180285059903
// MI455X (gfx1250) — matrix operand order repaired and checked by reading, not compiled, not run on hardware
//
#include <hip/hip_runtime.h>

// MHA for B=2, N=2048, D=1024, H=16, HD=64 on gfx1250 (MI455X).
// All matmuls use v_wmma_f32_16x16x32_bf16 (f32 accumulate).
// GEMMs: 32x64 wave tile, register double-buffering + global_prefetch.

#define BB 2
#define NN 2048
#define DD 1024
#define HH 16
#define HD 64
#define SCALE 0.125f // HD^-0.5

typedef __bf16 v8bf  __attribute__((ext_vector_type(8)));
typedef __bf16 v16bf __attribute__((ext_vector_type(16)));
typedef float  v8f   __attribute__((ext_vector_type(8)));

static __device__ __forceinline__ unsigned short f2bfu(float f) {
  union { float f; unsigned u; } in; in.f = f;
  unsigned u = in.u;
  u += 0x7FFFu + ((u >> 16) & 1u); // round to nearest even
  return (unsigned short)(u >> 16);
}

// A-matrix fragment (16x32 bf16, M = lane&15):
// lane holds K = {h16..h16+15}, h16 = (lane>>4)*16 -> two adjacent 16B chunks.
static __device__ __forceinline__ v16bf load_a_frag(const unsigned short* p) {
  v8bf lo = *reinterpret_cast<const v8bf*>(p);
  v8bf hi = *reinterpret_cast<const v8bf*>(p + 8);
  return __builtin_shufflevector(lo, hi, 0,1,2,3,4,5,6,7,8,9,10,11,12,13,14,15);
}
// B-matrix fragment (32x16 bf16, N = lane&15):
// lane holds K = {h16..h16+15}, h16 = (lane>>4)*16 -> one contiguous 32B chunk.
static __device__ __forceinline__ v16bf load_b_frag(const unsigned short* p) {
  return *reinterpret_cast<const v16bf*>(p);
}
static __device__ __forceinline__ v8f wmma_bf16(v16bf a, v16bf b, v8f c) {
  return __builtin_amdgcn_wmma_f32_16x16x32_bf16(false, a, false, b, (short)0, c,
                                                 false, false);
}

// ---------------- stage 1: precision conversion ----------------
__global__ void k_cvt_bf16(const float* __restrict__ src,
                           unsigned short* __restrict__ dst, int n) {
  int i = blockIdx.x * blockDim.x + threadIdx.x;
  if (i < n) dst[i] = f2bfu(src[i]);
}

// Wt[n][k] = bf16(W[k][n])  (coalesced writes)
__global__ void k_transpose_cvt(const float* __restrict__ W,
                                unsigned short* __restrict__ Wt) {
  int i = blockIdx.x * blockDim.x + threadIdx.x;
  int nrow = i >> 10, k = i & (DD - 1);
  Wt[i] = f2bfu(W[(size_t)k * DD + nrow]);
}

// ---------------- stage 2/4: GEMM  C = A @ Wt^T + bias ----------
// One wave computes a 32x64 tile (8 WMMAs per 32-deep K step).
// mode 0: bf16 dst[bh][n][hd] (Q,K)
// mode 1: bf16 dst[bh][hd][n] (V transposed)
// mode 2: f32  dst[m][d]      (final output)
__global__ __launch_bounds__(256)
void k_gemm(const unsigned short* __restrict__ A,
            const unsigned short* __restrict__ Wt,
            const float* __restrict__ bias,
            void* __restrict__ dstv, int mode) {
  const int lane = threadIdx.x & 31, hf = lane >> 4, l16 = lane & 15;
  const int wid = blockIdx.x * 8 + (threadIdx.x >> 5);
  const int mBase = (wid & 127) * 32;   // 128 M tiles of 32 rows
  const int nBase = (wid >> 7) * 64;    // 16  N tiles of 64 cols

  v8f acc[2][4] = {};
  const unsigned short* ar0  = A + (size_t)(mBase + l16) * DD + hf * 16;
  const unsigned short* ar1  = ar0 + (size_t)16 * DD;
  const unsigned short* brow = Wt + (size_t)(nBase + l16) * DD + hf * 16;

  v16bf a0 = load_a_frag(ar0);
  v16bf a1 = load_a_frag(ar1);
  for (int kk = 0; kk < DD; kk += 32) {
    // current B fragments in distinct registers -> partial loadcnt waits
    v16bf b0 = load_b_frag(brow + kk);
    v16bf b1 = load_b_frag(brow + (size_t)16 * DD + kk);
    v16bf b2 = load_b_frag(brow + (size_t)32 * DD + kk);
    v16bf b3 = load_b_frag(brow + (size_t)48 * DD + kk);
    // software pipeline: next A fragments + prefetch next B rows
    const int kn = (kk + 32) & (DD - 1); // wraps to 0 on last iter (harmless)
    v16bf a0n = load_a_frag(ar0 + kn);
    v16bf a1n = load_a_frag(ar1 + kn);
    __builtin_prefetch(brow + kn, 0, 3);                      // global_prefetch_b8
    __builtin_prefetch(brow + (size_t)16 * DD + kn, 0, 3);
    __builtin_prefetch(brow + (size_t)32 * DD + kn, 0, 3);
    __builtin_prefetch(brow + (size_t)48 * DD + kn, 0, 3);

    acc[0][0] = wmma_bf16(a0, b0, acc[0][0]);
    acc[1][0] = wmma_bf16(a1, b0, acc[1][0]);
    acc[0][1] = wmma_bf16(a0, b1, acc[0][1]);
    acc[1][1] = wmma_bf16(a1, b1, acc[1][1]);
    acc[0][2] = wmma_bf16(a0, b2, acc[0][2]);
    acc[1][2] = wmma_bf16(a1, b2, acc[1][2]);
    acc[0][3] = wmma_bf16(a0, b3, acc[0][3]);
    acc[1][3] = wmma_bf16(a1, b3, acc[1][3]);
    a0 = a0n;
    a1 = a1n;
  }

#pragma unroll
  for (int i = 0; i < 2; ++i) {
#pragma unroll
    for (int t = 0; t < 4; ++t) {
      int d = nBase + t * 16 + l16;
      float bvv = bias[d];
      int h = d >> 6, hd = d & (HD - 1);
#pragma unroll
      for (int r = 0; r < 8; ++r) {
        int mrow = mBase + i * 16 + hf * 8 + r;
        float val = acc[i][t][r] + bvv;
        if (mode == 2) {
          ((float*)dstv)[(size_t)mrow * DD + d] = val;
        } else {
          int b = mrow >> 11, n = mrow & (NN - 1);
          size_t off = (mode == 0)
              ? ((size_t)(b * HH + h) * NN + n) * HD + hd
              : ((size_t)(b * HH + h) * HD + hd) * NN + n;
          ((unsigned short*)dstv)[off] = f2bfu(val);
        }
      }
    }
  }
}

// ---------------- stage 3: flash attention ----------------------
// One wave owns a 16-query tile of one (b,h); loops over 32-key blocks with
// online softmax. P is staged through a per-wave LDS tile (same-wave DS ops
// are in-order, no block barrier needed). V fragments are loaded BEFORE the
// softmax VALU section so exp/max/shuffle work hides the V load latency.
__global__ __launch_bounds__(256)
void k_attn(const unsigned short* __restrict__ qg,
            const unsigned short* __restrict__ kg,
            const unsigned short* __restrict__ vg, // [bh][hd][n]
            unsigned short* __restrict__ ctx) {    // bf16 [B*N, D]
  __shared__ unsigned short ldsP[8 * 16 * 32];
  const int lane = threadIdx.x & 31, hf = lane >> 4, l16 = lane & 15;
  const int wid = blockIdx.x * 8 + (threadIdx.x >> 5);
  const int qTile = wid & 127, bh = wid >> 7;
  const unsigned short* qp = qg + ((size_t)bh * NN + qTile * 16) * HD;
  const unsigned short* kp = kg + (size_t)bh * NN * HD;
  const unsigned short* vp = vg + (size_t)bh * HD * NN;
  unsigned short* lp = ldsP + (threadIdx.x >> 5) * (16 * 32);

  v16bf qa = load_a_frag(qp + l16 * HD + hf * 16);       // hd 0..31
  v16bf qb = load_a_frag(qp + l16 * HD + 32 + hf * 16);  // hd 32..63

  float mrun[8], lrun[8];
  v8f o0 = {}, o1 = {}, o2 = {}, o3 = {};
#pragma unroll
  for (int r = 0; r < 8; ++r) { mrun[r] = -3.0e38f; lrun[r] = 0.0f; }

  for (int j = 0; j < NN; j += 32) {
    const unsigned short* kr0 = kp + (size_t)(j + l16) * HD + hf * 16;
    const unsigned short* kr1 = kp + (size_t)(j + 16 + l16) * HD + hf * 16;
    v16bf kb00 = load_b_frag(kr0);
    v16bf kb01 = load_b_frag(kr0 + 32);
    v16bf kb10 = load_b_frag(kr1);
    v16bf kb11 = load_b_frag(kr1 + 32);
    // V fragments for this block, hoisted ahead of softmax
    v16bf vf0 = load_b_frag(vp + (size_t)(0 * 16 + l16) * NN + j + hf * 16);
    v16bf vf1 = load_b_frag(vp + (size_t)(1 * 16 + l16) * NN + j + hf * 16);
    v16bf vf2 = load_b_frag(vp + (size_t)(2 * 16 + l16) * NN + j + hf * 16);
    v16bf vf3 = load_b_frag(vp + (size_t)(3 * 16 + l16) * NN + j + hf * 16);
    // prefetch next key/value block
    const int jn = (j + 32) & (NN - 1);
    __builtin_prefetch(kp + (size_t)(jn + l16) * HD + hf * 16, 0, 3);
    __builtin_prefetch(vp + (size_t)l16 * NN + jn + hf * 16, 0, 3);

    v8f s0 = {}, s1 = {};
    s0 = wmma_bf16(qa, kb00, s0);
    s0 = wmma_bf16(qb, kb01, s0);
    s1 = wmma_bf16(qa, kb10, s1);
    s1 = wmma_bf16(qb, kb11, s1);

#pragma unroll
    for (int r = 0; r < 8; ++r) {
      float a0 = s0[r] * SCALE, a1 = s1[r] * SCALE;
      float mx = fmaxf(a0, a1);                 // row-wise over 32 keys:
      mx = fmaxf(mx, __shfl_xor(mx, 1));        // xor 1/2/4/8 stays within the
      mx = fmaxf(mx, __shfl_xor(mx, 2));        // 16-lane half that shares a row
      mx = fmaxf(mx, __shfl_xor(mx, 4));
      mx = fmaxf(mx, __shfl_xor(mx, 8));
      float mnew = fmaxf(mrun[r], mx);
      float alpha = __expf(mrun[r] - mnew);
      mrun[r] = mnew;
      float p0 = __expf(a0 - mnew);
      float p1 = __expf(a1 - mnew);
      float rs = p0 + p1;
      rs += __shfl_xor(rs, 1);
      rs += __shfl_xor(rs, 2);
      rs += __shfl_xor(rs, 4);
      rs += __shfl_xor(rs, 8);
      lrun[r] = lrun[r] * alpha + rs;
      o0[r] *= alpha; o1[r] *= alpha; o2[r] *= alpha; o3[r] *= alpha;
      // C-layout -> LDS (row-major 16x32 bf16) for A-fragment reload
      lp[(hf * 8 + r) * 32 + l16] = f2bfu(p0);
      lp[(hf * 8 + r) * 32 + 16 + l16] = f2bfu(p1);
    }
    v16bf pf = load_a_frag(lp + l16 * 32 + hf * 16);
    o0 = wmma_bf16(pf, vf0, o0);
    o1 = wmma_bf16(pf, vf1, o1);
    o2 = wmma_bf16(pf, vf2, o2);
    o3 = wmma_bf16(pf, vf3, o3);
  }

  const int b = bh / HH, h = bh % HH;
#pragma unroll
  for (int r = 0; r < 8; ++r) {
    float inv = 1.0f / lrun[r];
    int n = qTile * 16 + hf * 8 + r;
    size_t rowOff = (size_t)(b * NN + n) * DD + h * HD;
    ctx[rowOff + 0 * 16 + l16] = f2bfu(o0[r] * inv);
    ctx[rowOff + 1 * 16 + l16] = f2bfu(o1[r] * inv);
    ctx[rowOff + 2 * 16 + l16] = f2bfu(o2[r] * inv);
    ctx[rowOff + 3 * 16 + l16] = f2bfu(o3[r] * inv);
  }
}

extern "C" void kernel_launch(void* const* d_in, const int* in_sizes, int n_in,
                              void* d_out, int out_size, void* d_ws, size_t ws_size,
                              hipStream_t stream) {
  (void)in_sizes; (void)n_in; (void)out_size; (void)ws_size;
  const float* x  = (const float*)d_in[0];
  const float* Wq = (const float*)d_in[1];
  const float* bq = (const float*)d_in[2];
  const float* Wk = (const float*)d_in[3];
  const float* bk = (const float*)d_in[4];
  const float* Wv = (const float*)d_in[5];
  const float* bv = (const float*)d_in[6];
  const float* Wo = (const float*)d_in[7];
  const float* bo = (const float*)d_in[8];
  float* out = (float*)d_out;

  // workspace layout (bf16 as ushort): total 48 MiB
  unsigned short* p = (unsigned short*)d_ws;
  const size_t ACT = (size_t)BB * NN * DD; // 4,194,304 elems
  unsigned short* x16 = p; p += ACT;
  unsigned short* wqT = p; p += (size_t)DD * DD;
  unsigned short* wkT = p; p += (size_t)DD * DD;
  unsigned short* wvT = p; p += (size_t)DD * DD;
  unsigned short* woT = p; p += (size_t)DD * DD;
  unsigned short* qh  = p; p += ACT;   // [bh][n][hd]
  unsigned short* kh  = p; p += ACT;   // [bh][n][hd]
  unsigned short* vtT = p; p += ACT;   // [bh][hd][n]
  unsigned short* ctx = p; p += ACT;   // [B*N][D]

  k_cvt_bf16<<<(int)(ACT / 256), 256, 0, stream>>>(x, x16, (int)ACT);
  k_transpose_cvt<<<(DD * DD) / 256, 256, 0, stream>>>(Wq, wqT);
  k_transpose_cvt<<<(DD * DD) / 256, 256, 0, stream>>>(Wk, wkT);
  k_transpose_cvt<<<(DD * DD) / 256, 256, 0, stream>>>(Wv, wvT);
  k_transpose_cvt<<<(DD * DD) / 256, 256, 0, stream>>>(Wo, woT);

  // 2048 waves each: 128 Mtiles(32) x 16 Ntiles(64) -> 256 blocks of 8 waves
  k_gemm<<<256, 256, 0, stream>>>(x16, wqT, bq, qh, 0);
  k_gemm<<<256, 256, 0, stream>>>(x16, wkT, bk, kh, 0);
  k_gemm<<<256, 256, 0, stream>>>(x16, wvT, bv, vtT, 1);

  k_attn<<<512, 256, 0, stream>>>(qh, kh, vtT, ctx);

  k_gemm<<<256, 256, 0, stream>>>(ctx, woT, bo, out, 2);
}
